// MLANtops_1675037245517
// MI455X (gfx1250) — matrix operand order repaired and checked by reading, not compiled, not run on hardware
//
#include <hip/hip_runtime.h>
#include <hip/hip_bf16.h>
#include <cmath>

// ---------------------------------------------------------------------------
// MLA forward for MI455X (gfx1250): bf16 WMMA (16x16x32) everywhere, fp32 accum.
// Pipeline:
//  1) q_lat = x @ wq_a^T                (GEMM)
//  2) kv    = x @ wkv_a^T               (GEMM)
//  3) q_nrm = rmsnorm(q_lat)
//  4) k_eff = [rmsnorm(kv_c) || rope(k_pe)]
//  5) q     = q_nrm @ wq_b^T            (GEMM)
//  6) rope q_pe in place inside q
//  7) K_head[b,h] = kv_c @ w_nope[h]^T  (batched GEMM, absorbed MLA)
//  8) fused flash attention: S = Q[192] . [K_head||k_pe]^T, softmax, O = P.kv_c
//  9) aout[h] = wkv_out[h] @ w_val[h]^T (batched GEMM)
// 10) out = aout @ wo^T                 (GEMM)
// All GEMMs: K % 32 == 0 and leading dims % 4 == 0 (float4 / b128 loads legal).
// ---------------------------------------------------------------------------

typedef __attribute__((ext_vector_type(16))) __bf16 v16bf;
typedef __attribute__((ext_vector_type(8)))  float  v8f;

union V16U { v16bf v; unsigned short u[16]; };

// Pack two floats to packed bf16: round-to-nearest (+0x8000, ties-away) then a
// single v_perm_b32 assembles {hi16(b), hi16(a)} — 3 VALU ops per pair, no
// half-register v_mov_b16 shuffle chains.
__device__ inline unsigned int pack2bf(float a, float b) {
  unsigned int ua = __builtin_bit_cast(unsigned int, a) + 0x8000u;
  unsigned int ub = __builtin_bit_cast(unsigned int, b) + 0x8000u;
  return __builtin_amdgcn_perm(ub, ua, 0x07060302u);
}

__device__ inline v8f v8zero() {
  v8f z;
#pragma unroll
  for (int e = 0; e < 8; ++e) z[e] = 0.0f;
  return z;
}

__device__ inline v8f wmma_bf16(v16bf a, v16bf b, v8f c) {
  return __builtin_amdgcn_wmma_f32_16x16x32_bf16(false, a, false, b, (short)0, c,
                                                 false, false);
}

// A fragment (16x32 bf16, row-major LDS tile, 'stride' in ushorts).
// Layout used here: lane half hh, row = lane&15; element e -> K = hh*16 + e
__device__ inline v16bf frag_a(const unsigned short* base, int stride, int lane) {
  int hh = lane >> 4, row = lane & 15;
  V16U t;
#pragma unroll
  for (int e = 0; e < 16; ++e) {
    int k = (hh << 4) + e;
    t.u[e] = base[row * stride + k];
  }
  return t.v;
}

// B fragment for C = A . B^T : LDS tile holds B rows (N-major), K contiguous.
// B layout used here: col = lane&15, K = hh*16 + e
__device__ inline v16bf frag_bT(const unsigned short* base, int stride, int lane) {
  int hh = lane >> 4, col = lane & 15;
  V16U t;
#pragma unroll
  for (int e = 0; e < 16; ++e) t.u[e] = base[col * stride + (hh << 4) + e];
  return t.v;
}

// B fragment for non-transposed B[K][N] tile (used for P @ V).
__device__ inline v16bf frag_bN(const unsigned short* base, int stride, int lane) {
  int hh = lane >> 4, col = lane & 15;
  V16U t;
#pragma unroll
  for (int e = 0; e < 16; ++e) t.u[e] = base[((hh << 4) + e) * stride + col];
  return t.v;
}

// ---------------------------------------------------------------------------
// Generic batched GEMM: C = A(M,K) . B(N,K)^T, fp32 in/out, bf16 WMMA compute.
// Requirements: K % 32 == 0, lda/ldb % 4 == 0, base pointers 16B aligned.
// z = blockIdx.z ; z1 = z / zdiv, z2 = z % zdiv ; base += z1*s?1 + z2*s?2.
// Block = 256 thr (8 waves). Block tile 128x128, K-step 32, double-buffered LDS.
// Wave grid 4(M) x 2(N): each wave computes 32x64 = 2x4 wmma tiles.
// ---------------------------------------------------------------------------
__global__ __launch_bounds__(256) void gemm_bt_kernel(
    const float* __restrict__ A, const float* __restrict__ B,
    float* __restrict__ C, int M, int N, int K, int lda, int ldb, int ldc,
    long long sA1, long long sA2, long long sB1, long long sB2,
    long long sC1, long long sC2, int zdiv) {
  __shared__ unsigned int As[2][128][20];   // 2 x 128 x 32 bf16 (+8 ushort pad)
  __shared__ unsigned int Bs[2][128][20];

  int z = blockIdx.z, z1 = z / zdiv, z2 = z % zdiv;
  const float* Ab = A + z1 * sA1 + z2 * sA2;
  const float* Bb = B + z1 * sB1 + z2 * sB2;
  float* Cb = C + z1 * sC1 + z2 * sC2;

  int m0 = blockIdx.y * 128, n0 = blockIdx.x * 128;
  int tid = threadIdx.x, lane = tid & 31, wave = tid >> 5;
  int hh = lane >> 4, r16 = lane & 15;
  int rowbase = (wave & 3) * 32;
  int colbase = (wave >> 2) * 64;

  v8f acc[2][4];
#pragma unroll
  for (int i = 0; i < 2; ++i)
#pragma unroll
    for (int j = 0; j < 4; ++j) acc[i][j] = v8zero();

  int lrow = tid >> 1;            // 0..127: tile row this thread loads
  int lkoff = (tid & 1) * 16;     // 0 or 16: K offset this thread loads
  bool mok = (m0 + lrow) < M;
  bool nok = (n0 + lrow) < N;
  const float* Arow = Ab + (long long)(m0 + lrow) * lda + lkoff;
  const float* Brow = Bb + (long long)(n0 + lrow) * ldb + lkoff;

  float4 ra0, ra1, ra2, ra3, rb0, rb1, rb2, rb3;
  const float4 zf4 = {0.0f, 0.0f, 0.0f, 0.0f};

  // prefetch stage 0
  if (mok) {
    const float4* p = (const float4*)Arow;
    ra0 = p[0]; ra1 = p[1]; ra2 = p[2]; ra3 = p[3];
  } else { ra0 = ra1 = ra2 = ra3 = zf4; }
  if (nok) {
    const float4* p = (const float4*)Brow;
    rb0 = p[0]; rb1 = p[1]; rb2 = p[2]; rb3 = p[3];
  } else { rb0 = rb1 = rb2 = rb3 = zf4; }
  {
    unsigned int* ap = &As[0][lrow][lkoff >> 1];
    ap[0] = pack2bf(ra0.x, ra0.y); ap[1] = pack2bf(ra0.z, ra0.w);
    ap[2] = pack2bf(ra1.x, ra1.y); ap[3] = pack2bf(ra1.z, ra1.w);
    ap[4] = pack2bf(ra2.x, ra2.y); ap[5] = pack2bf(ra2.z, ra2.w);
    ap[6] = pack2bf(ra3.x, ra3.y); ap[7] = pack2bf(ra3.z, ra3.w);
    unsigned int* bp = &Bs[0][lrow][lkoff >> 1];
    bp[0] = pack2bf(rb0.x, rb0.y); bp[1] = pack2bf(rb0.z, rb0.w);
    bp[2] = pack2bf(rb1.x, rb1.y); bp[3] = pack2bf(rb1.z, rb1.w);
    bp[4] = pack2bf(rb2.x, rb2.y); bp[5] = pack2bf(rb2.z, rb2.w);
    bp[6] = pack2bf(rb3.x, rb3.y); bp[7] = pack2bf(rb3.z, rb3.w);
  }

  int nst = K >> 5;
  for (int s = 0; s < nst; ++s) {
    __syncthreads();
    int buf = s & 1;
    bool more = (s + 1) < nst;
    if (more) {                       // issue next-stage global loads early
      int koff = (s + 1) << 5;
      if (mok) {
        const float4* p = (const float4*)(Arow + koff);
        ra0 = p[0]; ra1 = p[1]; ra2 = p[2]; ra3 = p[3];
      }
      if (nok) {
        const float4* p = (const float4*)(Brow + koff);
        rb0 = p[0]; rb1 = p[1]; rb2 = p[2]; rb3 = p[3];
      }
    }

    const unsigned short* A16 = (const unsigned short*)&As[buf][0][0];
    const unsigned short* B16 = (const unsigned short*)&Bs[buf][0][0];
    v16bf af[2], bfr[4];
#pragma unroll
    for (int i = 0; i < 2; ++i) af[i] = frag_a(A16 + (rowbase + i * 16) * 40, 40, lane);
#pragma unroll
    for (int j = 0; j < 4; ++j) bfr[j] = frag_bT(B16 + (colbase + j * 16) * 40, 40, lane);
#pragma unroll
    for (int i = 0; i < 2; ++i)
#pragma unroll
      for (int j = 0; j < 4; ++j) acc[i][j] = wmma_bf16(af[i], bfr[j], acc[i][j]);

    if (more) {                       // store next stage into the other buffer
      unsigned int* ap = &As[buf ^ 1][lrow][lkoff >> 1];
      ap[0] = pack2bf(ra0.x, ra0.y); ap[1] = pack2bf(ra0.z, ra0.w);
      ap[2] = pack2bf(ra1.x, ra1.y); ap[3] = pack2bf(ra1.z, ra1.w);
      ap[4] = pack2bf(ra2.x, ra2.y); ap[5] = pack2bf(ra2.z, ra2.w);
      ap[6] = pack2bf(ra3.x, ra3.y); ap[7] = pack2bf(ra3.z, ra3.w);
      unsigned int* bp = &Bs[buf ^ 1][lrow][lkoff >> 1];
      bp[0] = pack2bf(rb0.x, rb0.y); bp[1] = pack2bf(rb0.z, rb0.w);
      bp[2] = pack2bf(rb1.x, rb1.y); bp[3] = pack2bf(rb1.z, rb1.w);
      bp[4] = pack2bf(rb2.x, rb2.y); bp[5] = pack2bf(rb2.z, rb2.w);
      bp[6] = pack2bf(rb3.x, rb3.y); bp[7] = pack2bf(rb3.z, rb3.w);
    }
  }

#pragma unroll
  for (int i = 0; i < 2; ++i)
#pragma unroll
    for (int j = 0; j < 4; ++j)
#pragma unroll
      for (int rr = 0; rr < 8; ++rr) {
        int m = m0 + rowbase + i * 16 + rr + hh * 8;
        int n = n0 + colbase + j * 16 + r16;
        if (m < M && n < N) Cb[(long long)m * ldc + n] = acc[i][j][rr];
      }
}

// ---------------------------------------------------------------------------
// RMS norm over rows of width W (one block per row).
// ---------------------------------------------------------------------------
__global__ __launch_bounds__(256) void rmsnorm_kernel(
    const float* __restrict__ x, const float* __restrict__ w,
    float* __restrict__ y, int W) {
  __shared__ float red[256];
  int tid = threadIdx.x;
  const float* xr = x + (long long)blockIdx.x * W;
  float s = 0.0f;
  for (int i = tid; i < W; i += 256) { float v = xr[i]; s += v * v; }
  red[tid] = s;
  __syncthreads();
  for (int off = 128; off > 0; off >>= 1) {
    if (tid < off) red[tid] += red[tid + off];
    __syncthreads();
  }
  float sc = rsqrtf(red[0] / (float)W + 1e-6f);
  float* yr = y + (long long)blockIdx.x * W;
  for (int i = tid; i < W; i += 256) yr[i] = xr[i] * sc * w[i];
}

// ---------------------------------------------------------------------------
// kv row (576) -> k_eff row: [rmsnorm(kv_c, w) (512) || rope(k_pe) (64)]
// ---------------------------------------------------------------------------
__global__ __launch_bounds__(256) void kv_prep_kernel(
    const float* __restrict__ kv, const float* __restrict__ w,
    const float* __restrict__ cos_t, const float* __restrict__ sin_t,
    float* __restrict__ k_eff) {
  __shared__ float red[256];
  int tid = threadIdx.x;
  int row = blockIdx.x;                  // bs index, s = row % 512
  const float* xr = kv + (long long)row * 576;
  float v0 = xr[tid], v1 = xr[256 + tid];
  red[tid] = v0 * v0 + v1 * v1;
  __syncthreads();
  for (int off = 128; off > 0; off >>= 1) {
    if (tid < off) red[tid] += red[tid + off];
    __syncthreads();
  }
  float sc = rsqrtf(red[0] / 512.0f + 1e-6f);
  float* out = k_eff + (long long)row * 576;
  out[tid] = v0 * sc * w[tid];
  out[256 + tid] = v1 * sc * w[256 + tid];
  if (tid < 32) {
    int s = row & 511, j = tid;
    float xre = xr[512 + 2 * j], xim = xr[512 + 2 * j + 1];
    float c = cos_t[s * 32 + j], si = sin_t[s * 32 + j];
    out[512 + 2 * j]     = xre * c - xim * si;
    out[512 + 2 * j + 1] = xre * si + xim * c;
  }
}

// ---------------------------------------------------------------------------
// Rope q_pe in place: q layout (1024, 128 heads, 192), pe = cols 128..191.
// One thread per (bs, h, pair j).
// ---------------------------------------------------------------------------
__global__ __launch_bounds__(256) void q_rope_kernel(
    float* __restrict__ q, const float* __restrict__ cos_t,
    const float* __restrict__ sin_t) {
  int gid = blockIdx.x * 256 + threadIdx.x;    // 1024*128*32 total
  int j = gid & 31;
  int h = (gid >> 5) & 127;
  int bs = gid >> 12;
  int s = bs & 511;
  long long base = (long long)bs * 24576 + h * 192 + 128;
  float xre = q[base + 2 * j], xim = q[base + 2 * j + 1];
  float c = cos_t[s * 32 + j], si = sin_t[s * 32 + j];
  q[base + 2 * j]     = xre * c - xim * si;
  q[base + 2 * j + 1] = xre * si + xim * c;
}

// ---------------------------------------------------------------------------
// Fused flash attention (absorbed MLA).
// Grid: (S/32, H, B). Block 256 thr (8 waves). Query tile 32, key tile 32.
// Q (32x192) from q;  K tile = [K_head (128) || k_pe (64)];  V = kv_c (512).
// Online softmax in LDS; O accum in registers (wave w owns V-cols w*64..w*64+63).
// Writes wkv_out (bs, h, 512) fp32.
// ---------------------------------------------------------------------------
__global__ __launch_bounds__(256) void attn_kernel(
    const float* __restrict__ q,        // (1024, 128, 192) rope applied
    const float* __restrict__ k_head,   // (B*H, 512, 128)
    const float* __restrict__ k_eff,    // (1024, 576): [0,512)=kv_c, [512,576)=k_pe
    float* __restrict__ wkv_o,          // (1024, 128, 512)
    float scale) {
  __shared__ unsigned int Qs[32][96];   // 32 x 192 bf16
  __shared__ unsigned int Ks[32][96];   // 32 x 192 bf16
  __shared__ unsigned int Vs[32][256];  // 32 x 512 bf16
  __shared__ unsigned int Ps[32][18];   // 32 x 32 bf16 (+pad)
  __shared__ float Ss[32][33];
  __shared__ float sm_m[32], sm_l[32], sm_r[32];

  int s0 = blockIdx.x * 32;
  int h = blockIdx.y;
  int b = blockIdx.z;
  int tid = threadIdx.x, lane = tid & 31, wave = tid >> 5;
  int hh = lane >> 4, r16 = lane & 15;
  int colbase = wave * 64;                 // this wave's V-column slice

  // Q tile: 32 rows x 192 cols as float4 loads, packed bf16 stores
  for (int i = tid; i < 32 * 48; i += 256) {
    int rr = i / 48, c4 = (i % 48) * 4;
    const float4 v = *(const float4*)(q + (long long)(b * 512 + s0 + rr) * 24576 +
                                      h * 192 + c4);
    Qs[rr][c4 >> 1] = pack2bf(v.x, v.y);
    Qs[rr][(c4 >> 1) + 1] = pack2bf(v.z, v.w);
  }
  if (tid < 32) { sm_m[tid] = -1e30f; sm_l[tid] = 0.0f; }

  v8f acc[2][4];
#pragma unroll
  for (int i = 0; i < 2; ++i)
#pragma unroll
    for (int j = 0; j < 4; ++j) acc[i][j] = v8zero();

  int ntile = (s0 >> 5) + 1;               // causal: only t0 <= s0
  for (int kt = 0; kt < ntile; ++kt) {
    int t0 = kt * 32;
    __syncthreads();
    for (int i = tid; i < 32 * 48; i += 256) {
      int rr = i / 48, c4 = (i % 48) * 4;
      float4 v;
      if (c4 < 128)
        v = *(const float4*)(k_head + ((long long)(b * 128 + h) * 512 + t0 + rr) * 128 + c4);
      else
        v = *(const float4*)(k_eff + (long long)(b * 512 + t0 + rr) * 576 + 512 + (c4 - 128));
      Ks[rr][c4 >> 1] = pack2bf(v.x, v.y);
      Ks[rr][(c4 >> 1) + 1] = pack2bf(v.z, v.w);
    }
    for (int i = tid; i < 32 * 128; i += 256) {
      int rr = i >> 7, c4 = (i & 127) * 4;
      const float4 v = *(const float4*)(k_eff + (long long)(b * 512 + t0 + rr) * 576 + c4);
      Vs[rr][c4 >> 1] = pack2bf(v.x, v.y);
      Vs[rr][(c4 >> 1) + 1] = pack2bf(v.z, v.w);
    }
    __syncthreads();

    // S = Q . K^T : 2x2 tiles of 16x16, waves 0..3
    if (wave < 4) {
      int ti = wave >> 1, tj = wave & 1;
      const unsigned short* Q16 = (const unsigned short*)&Qs[0][0];
      const unsigned short* K16 = (const unsigned short*)&Ks[0][0];
      v8f sa = v8zero();
#pragma unroll
      for (int kk = 0; kk < 192; kk += 32) {
        v16bf af = frag_a(Q16 + (ti * 16) * 192 + kk, 192, lane);
        v16bf bb = frag_bT(K16 + (tj * 16) * 192 + kk, 192, lane);
        sa = wmma_bf16(af, bb, sa);
      }
#pragma unroll
      for (int rr = 0; rr < 8; ++rr) {
        int m = ti * 16 + rr + hh * 8;
        int n = tj * 16 + r16;
        int sg = s0 + m, tg = t0 + n;
        Ss[m][n] = (tg <= sg) ? sa[rr] * scale : -1e30f;
      }
    }
    __syncthreads();

    // Online softmax row update (thread r handles row r)
    if (tid < 32) {
      int m = tid;
      float mo = sm_m[m], mx = mo;
      for (int n = 0; n < 32; ++n) mx = fmaxf(mx, Ss[m][n]);
      float rs = __expf(mo - mx);
      float sum = 0.0f;
      for (int n = 0; n < 32; n += 2) {
        float p0 = __expf(Ss[m][n] - mx);
        float p1 = __expf(Ss[m][n + 1] - mx);
        sum += p0 + p1;
        Ps[m][n >> 1] = pack2bf(p0, p1);
      }
      sm_m[m] = mx;
      sm_l[m] = rs * sm_l[m] + sum;
      sm_r[m] = rs;
    }
    __syncthreads();

    // Rescale O accumulators, then O += P . V
#pragma unroll
    for (int i = 0; i < 2; ++i)
#pragma unroll
      for (int j = 0; j < 4; ++j)
#pragma unroll
        for (int rr = 0; rr < 8; ++rr)
          acc[i][j][rr] *= sm_r[i * 16 + rr + hh * 8];

    const unsigned short* P16 = (const unsigned short*)&Ps[0][0];
    const unsigned short* V16p = (const unsigned short*)&Vs[0][0];
    v16bf pf[2];
#pragma unroll
    for (int i = 0; i < 2; ++i) pf[i] = frag_a(P16 + (i * 16) * 36, 36, lane);
#pragma unroll
    for (int j = 0; j < 4; ++j) {
      v16bf vf = frag_bN(V16p + colbase + j * 16, 512, lane);
#pragma unroll
      for (int i = 0; i < 2; ++i) acc[i][j] = wmma_bf16(pf[i], vf, acc[i][j]);
    }
  }

  // Epilogue: divide by softmax denominator, store wkv_out
#pragma unroll
  for (int i = 0; i < 2; ++i)
#pragma unroll
    for (int j = 0; j < 4; ++j)
#pragma unroll
      for (int rr = 0; rr < 8; ++rr) {
        int m = i * 16 + rr + hh * 8;
        int n = colbase + j * 16 + r16;
        float o = acc[i][j][rr] / sm_l[m];
        wkv_o[(long long)(b * 512 + s0 + m) * 65536 + h * 512 + n] = o;
      }
}

// ---------------------------------------------------------------------------
extern "C" void kernel_launch(void* const* d_in, const int* in_sizes, int n_in,
                              void* d_out, int out_size, void* d_ws,
                              size_t ws_size, hipStream_t stream) {
  const float* x        = (const float*)d_in[0];
  const float* wq_a     = (const float*)d_in[1];
  const float* q_norm_w = (const float*)d_in[2];
  const float* wq_b     = (const float*)d_in[3];
  const float* wkv_a    = (const float*)d_in[4];
  const float* kv_norm_w= (const float*)d_in[5];
  const float* wkv_b    = (const float*)d_in[6];
  const float* wo       = (const float*)d_in[7];
  const float* fcos     = (const float*)d_in[8];
  const float* fsin     = (const float*)d_in[9];
  // d_in[10] mask (causal computed in-kernel), d_in[11] start_pos (== 0)

  float* ws = (float*)d_ws;
  size_t o = 0;
  float* q_lat = ws + o; o += (size_t)1024 * 1536;
  float* q_nrm = ws + o; o += (size_t)1024 * 1536;
  float* kvbuf = ws + o; o += (size_t)1024 * 576;
  float* k_eff = ws + o; o += (size_t)1024 * 576;
  float* qbuf  = ws + o; o += (size_t)1024 * 24576;
  float* khead = ws + o; o += (size_t)256 * 512 * 128;
  float* wkvo  = ws + o; o += (size_t)1024 * 128 * 512;
  float* aout  = ws + o; o += (size_t)1024 * 16384;
  (void)ws_size; (void)in_sizes; (void)n_in; (void)out_size;

  dim3 blk(256);

  // 1) q_lat = x @ wq_a^T   (1024x1536, K=7168)
  gemm_bt_kernel<<<dim3(12, 8, 1), blk, 0, stream>>>(
      x, wq_a, q_lat, 1024, 1536, 7168, 7168, 7168, 1536,
      0, 0, 0, 0, 0, 0, 1);
  // 2) kv = x @ wkv_a^T     (1024x576, K=7168)
  gemm_bt_kernel<<<dim3(5, 8, 1), blk, 0, stream>>>(
      x, wkv_a, kvbuf, 1024, 576, 7168, 7168, 7168, 576,
      0, 0, 0, 0, 0, 0, 1);
  // 3) q_nrm = rmsnorm(q_lat)
  rmsnorm_kernel<<<dim3(1024), blk, 0, stream>>>(q_lat, q_norm_w, q_nrm, 1536);
  // 4) k_eff = [rmsnorm(kv_c) || rope(k_pe)]
  kv_prep_kernel<<<dim3(1024), blk, 0, stream>>>(kvbuf, kv_norm_w, fcos, fsin, k_eff);
  // 5) q = q_nrm @ wq_b^T   (1024x24576, K=1536)
  gemm_bt_kernel<<<dim3(192, 8, 1), blk, 0, stream>>>(
      q_nrm, wq_b, qbuf, 1024, 24576, 1536, 1536, 1536, 24576,
      0, 0, 0, 0, 0, 0, 1);
  // 6) rope q_pe in place
  q_rope_kernel<<<dim3(16384), blk, 0, stream>>>(qbuf, fcos, fsin);
  // 7) K_head[b,h] = kv_c @ w_nope[h]^T  (z = b*128+h; 512x128, K=512)
  gemm_bt_kernel<<<dim3(1, 4, 256), blk, 0, stream>>>(
      k_eff, wkv_b, khead, 512, 128, 512, 576, 512, 128,
      (long long)512 * 576, 0,                       // A: +b*S*576
      0, (long long)256 * 512,                       // B: +h*256*512
      (long long)128 * 512 * 128, (long long)512 * 128,  // C: +b*H*T*128 + h*T*128
      128);
  // 8) fused attention -> wkv_out
  float mscale = 0.1f * logf(40.0f) + 1.0f;
  float scale = (1.0f / sqrtf(192.0f)) * mscale * mscale;
  attn_kernel<<<dim3(16, 128, 2), blk, 0, stream>>>(qbuf, khead, k_eff, wkvo, scale);
  // 9) aout[:,h,:] = wkv_out[:,h,:] @ w_val[h]^T  (z = h; 1024x128, K=512)
  gemm_bt_kernel<<<dim3(1, 8, 128), blk, 0, stream>>>(
      wkvo, wkv_b + 128 * 512, aout, 1024, 128, 512, 65536, 512, 16384,
      0, 512,                                        // A: +h*512 (lda=H*512)
      0, (long long)256 * 512,                       // B: +h*256*512 (rows 128..255)
      0, 128,                                        // C: +h*128 (ldc=H*128)
      128);
  // 10) out = aout @ wo^T   (1024x7168, K=16384)
  gemm_bt_kernel<<<dim3(56, 8, 1), blk, 0, stream>>>(
      aout, wo, (float*)d_out, 1024, 7168, 16384, 16384, 16384, 7168,
      0, 0, 0, 0, 0, 0, 1);
}
